// SelectiveStateSpace_71021579207189
// MI455X (gfx1250) — compile-verified, matrix operand order checked by reading, not run on hardware
//
#include <hip/hip_runtime.h>

// ---------------------------------------------------------------------------
// Mamba-style selective SSM for MI455X (gfx1250, wave32, WMMA).
// GEMMs: v_wmma_f32_16x16x32_f16 (f16 in, f32 accum), 32x64 macro-tile/wave,
// fully unrolled K loop (templated) for branch-free scheduling.
// Scan: 3-pass chunked diagonal linear scan (chunk=64, 216 chunks).
// ---------------------------------------------------------------------------

typedef __attribute__((ext_vector_type(16))) _Float16 v16h;
typedef __attribute__((ext_vector_type(8)))  _Float16 v8h;
typedef __attribute__((ext_vector_type(8)))  float    v8f;

#define DIMC     128
#define D_STATE  16
#define D_INNER  256
#define DT_RANK  8
#define LSEQ     13824          // 24*24*24
#define NBATCH   2
#define NROW     (NBATCH*LSEQ)  // 27648
#define XCOLS    48             // DT_RANK + 2*D_STATE = 40, padded to 48
#define CHUNK    64
#define NCHUNK   (LSEQ/CHUNK)   // 216

union FragU { v16h v; v8h h[2]; };

__device__ __forceinline__ v16h load_frag(const _Float16* p) {
  // CDNA5 16-bit A/B fragment: per lane, halves 0..7 <- K = g*8..g*8+7,
  // halves 8..15 <- K = 16+g*8..16+g*8+7 (two contiguous 16B runs).
  FragU u;
  u.h[0] = *(const v8h*)(p);
  u.h[1] = *(const v8h*)(p + 16);
  return u.v;
}

// ---------------- weight conversion (once per launch, tiny) ----------------
__global__ void cvt_f16_kernel(const float* __restrict__ src,
                               _Float16* __restrict__ dst, int n) {
  int i = blockIdx.x * 256 + threadIdx.x;
  if (i < n) dst[i] = (_Float16)src[i];
}

__global__ void pack_wx_kernel(const float* __restrict__ wx,
                               _Float16* __restrict__ dst) {
  int i = blockIdx.x * 256 + threadIdx.x;
  if (i >= XCOLS * D_INNER) return;
  int n = i / D_INNER, k = i % D_INNER;
  dst[i] = (n < DT_RANK + 2 * D_STATE) ? (_Float16)wx[n * D_INNER + k]
                                       : (_Float16)0.0f;
}

// ---------------- layernorm over C, (B,C,L) -> f16 (B*L, 128) --------------
__global__ void ln_kernel(const float* __restrict__ x,
                          const float* __restrict__ w,
                          const float* __restrict__ bsh,
                          _Float16* __restrict__ xn) {
  __shared__ float tile[32 * 129];   // stride 129 -> conflict-free banks
  __shared__ float mu[32], rs[32];
  const int b   = blockIdx.y;
  const int l0  = blockIdx.x * 32;
  const int tid = threadIdx.x;
  for (int idx = tid; idx < 32 * DIMC; idx += 256) {
    int c = idx >> 5, lof = idx & 31;               // coalesced along L
    tile[lof * 129 + c] =
        x[(size_t)b * DIMC * LSEQ + (size_t)c * LSEQ + l0 + lof];
  }
  __syncthreads();
  if (tid < 32) {
    float s = 0.f, s2 = 0.f;
    for (int c = 0; c < DIMC; ++c) {
      float v = tile[tid * 129 + c];
      s += v; s2 += v * v;
    }
    float m = s * (1.0f / DIMC);
    float var = s2 * (1.0f / DIMC) - m * m;
    mu[tid] = m;
    rs[tid] = rsqrtf(var + 1e-5f);
  }
  __syncthreads();
  for (int idx = tid; idx < 32 * DIMC; idx += 256) {
    int lof = idx >> 7, c = idx & 127;              // coalesced along C
    float v = (tile[lof * 129 + c] - mu[lof]) * rs[lof] * w[c] + bsh[c];
    xn[((size_t)(b * LSEQ + l0 + lof)) * DIMC + c] = (_Float16)v;
  }
}

// ---------------- WMMA GEMM: C[M,N] = A[M,K] * Bt[N,K]^T -------------------
// MT x NT tiles of 16x16 per wave; K compile-time => fully unrolled,
// branch-free inner loop so the scheduler can clause loads + overlap WMMAs.
// TRANS=false: row-major C[M,N].  TRANS=true: out[b, n, l] with m=(b,l).
template <int MT, int NT, int KK, bool TRANS>
__global__ void gemm_wmma(const _Float16* __restrict__ A,
                          const _Float16* __restrict__ Bt,
                          float* __restrict__ C,
                          int M, int N, int wavesPerRow) {
  const int wave = (blockIdx.x * blockDim.x + threadIdx.x) >> 5;
  const int lane = threadIdx.x & 31;
  const int m0 = (wave / wavesPerRow) * (16 * MT);
  const int n0 = (wave % wavesPerRow) * (16 * NT);
  if (m0 >= M) return;
  const int r = lane & 15, g = lane >> 4;

  const _Float16* aptr[MT];
#pragma unroll
  for (int mt = 0; mt < MT; ++mt)
    aptr[mt] = A + (size_t)(m0 + mt * 16 + r) * KK + g * 8;
  const _Float16* bptr[NT];
#pragma unroll
  for (int nt = 0; nt < NT; ++nt)
    bptr[nt] = Bt + (size_t)(n0 + nt * 16 + r) * KK + g * 8;

  v8f acc[MT][NT];
#pragma unroll
  for (int mt = 0; mt < MT; ++mt)
#pragma unroll
    for (int nt = 0; nt < NT; ++nt) acc[mt][nt] = (v8f){};

#pragma unroll
  for (int k0 = 0; k0 < KK; k0 += 32) {
    v16h af[MT], bf[NT];
#pragma unroll
    for (int mt = 0; mt < MT; ++mt) af[mt] = load_frag(aptr[mt] + k0);
#pragma unroll
    for (int nt = 0; nt < NT; ++nt) bf[nt] = load_frag(bptr[nt] + k0);
#pragma unroll
    for (int mt = 0; mt < MT; ++mt)
#pragma unroll
      for (int nt = 0; nt < NT; ++nt)
        acc[mt][nt] = __builtin_amdgcn_wmma_f32_16x16x32_f16(
            false, af[mt], false, bf[nt], (short)0, acc[mt][nt], false, false);
  }

#pragma unroll
  for (int mt = 0; mt < MT; ++mt) {
#pragma unroll
    for (int nt = 0; nt < NT; ++nt) {
      int n = n0 + nt * 16 + r;
#pragma unroll
      for (int i = 0; i < 8; ++i) {
        int m = m0 + mt * 16 + g * 8 + i;  // D layout: vgpr i -> M (+8 if g)
        if (TRANS) {
          int b = m / LSEQ, l = m % LSEQ;
          C[((size_t)b * DIMC + n) * LSEQ + l] = acc[mt][nt][i];
        } else {
          C[(size_t)m * N + n] = acc[mt][nt][i];
        }
      }
    }
  }
}

// ---------------- depthwise causal conv (k=4) + bias + SiLU ----------------
__global__ void conv_silu_kernel(const float* __restrict__ xz,
                                 const float* __restrict__ cw,
                                 const float* __restrict__ cb,
                                 float* __restrict__ uf,
                                 _Float16* __restrict__ uh) {
  int idx = blockIdx.x * 256 + threadIdx.x;
  if (idx >= NROW * D_INNER) return;
  int d = idx & 255;
  int row = idx >> 8;
  int l = row % LSEQ;
  float acc = cb[d];
#pragma unroll
  for (int j = 0; j < 4; ++j) {
    int ls = l - 3 + j;
    if (ls >= 0)
      acc += xz[(size_t)(row - 3 + j) * (2 * D_INNER) + d] * cw[d * 4 + j];
  }
  float sv = acc / (1.0f + __expf(-acc));   // SiLU
  uf[idx] = sv;
  uh[idx] = (_Float16)sv;
}

// ---------------- delta = softplus(dt_r @ W_dt^T + b_dt) -------------------
__global__ void delta_kernel(const float* __restrict__ xdbl,
                             const float* __restrict__ wdt,
                             const float* __restrict__ bdt,
                             float* __restrict__ delta) {
  int idx = blockIdx.x * 256 + threadIdx.x;
  if (idx >= NROW * D_INNER) return;
  int d = idx & 255;
  int row = idx >> 8;
  const float* xr = xdbl + (size_t)row * XCOLS;
  float acc = bdt[d];
#pragma unroll
  for (int rr = 0; rr < DT_RANK; ++rr) acc += xr[rr] * wdt[d * DT_RANK + rr];
  delta[idx] = (acc > 20.0f) ? acc : log1pf(__expf(acc));
}

// ---------------- scan pass 1: per-chunk (P = prod a, Q = local scan) ------
__global__ void scan_pass1(const float* __restrict__ delta,
                           const float* __restrict__ xdbl,
                           const float* __restrict__ uf,
                           const float* __restrict__ A_log,
                           float* __restrict__ Pb,
                           float* __restrict__ Qb) {
  int tid = threadIdx.x;
  int s = tid & 15, dl = tid >> 4;       // 16 states of one d in one half-wave
  int d = blockIdx.z * 16 + dl;
  int b = blockIdx.y;
  int chunk = blockIdx.x;
  float Av = -__expf(A_log[d * D_STATE + s]);
  float P = 1.0f, Q = 0.0f;
  int t0 = chunk * CHUNK;
  for (int t = 0; t < CHUNK; ++t) {
    int row = b * LSEQ + t0 + t;
    float dt = delta[(size_t)row * D_INNER + d];
    float a = __expf(dt * Av);
    float bs = xdbl[(size_t)row * XCOLS + DT_RANK + s];
    float uu = uf[(size_t)row * D_INNER + d];
    Q = a * Q + dt * uu * bs;
    P *= a;
  }
  size_t o = (((size_t)chunk * NBATCH + b) * D_INNER + d) * D_STATE + s;
  Pb[o] = P;
  Qb[o] = Q;
}

// ---------------- scan pass 2: sequential combine across chunks ------------
__global__ void scan_pass2(const float* __restrict__ Pb,
                           const float* __restrict__ Qb,
                           float* __restrict__ Hin) {
  int tid = threadIdx.x;
  int s = tid & 15, dl = tid >> 4;
  int d = blockIdx.y * 16 + dl;
  int b = blockIdx.x;
  float h = 0.0f;
  for (int chunk = 0; chunk < NCHUNK; ++chunk) {
    size_t o = (((size_t)chunk * NBATCH + b) * D_INNER + d) * D_STATE + s;
    Hin[o] = h;                       // exclusive prefix: state entering chunk
    h = Pb[o] * h + Qb[o];
  }
}

// ---------------- scan pass 3: replay + y = h.C + D*u, gate, f16 -----------
__global__ void scan_pass3(const float* __restrict__ delta,
                           const float* __restrict__ xdbl,
                           const float* __restrict__ uf,
                           const float* __restrict__ xz,
                           const float* __restrict__ A_log,
                           const float* __restrict__ Dp,
                           const float* __restrict__ Hin,
                           _Float16* __restrict__ yg) {
  int tid = threadIdx.x;
  int s = tid & 15, dl = tid >> 4;
  int d = blockIdx.z * 16 + dl;
  int b = blockIdx.y;
  int chunk = blockIdx.x;
  float Av = -__expf(A_log[d * D_STATE + s]);
  size_t o = (((size_t)chunk * NBATCH + b) * D_INNER + d) * D_STATE + s;
  float h = Hin[o];
  float Dd = Dp[d];
  int t0 = chunk * CHUNK;
  for (int t = 0; t < CHUNK; ++t) {
    int row = b * LSEQ + t0 + t;
    float dt = delta[(size_t)row * D_INNER + d];
    float a = __expf(dt * Av);
    float bs = xdbl[(size_t)row * XCOLS + DT_RANK + s];
    float uu = uf[(size_t)row * D_INNER + d];
    h = a * h + dt * uu * bs;
    float cs = xdbl[(size_t)row * XCOLS + DT_RANK + D_STATE + s];
    float yc = h * cs;
    // reduce over 16 states: consecutive lanes within wave32 half-wave
    yc += __shfl_xor(yc, 1, 16);
    yc += __shfl_xor(yc, 2, 16);
    yc += __shfl_xor(yc, 4, 16);
    yc += __shfl_xor(yc, 8, 16);
    if (s == 0) {
      float y = yc + Dd * uu;
      float z = xz[(size_t)row * (2 * D_INNER) + D_INNER + d];
      float zg = z / (1.0f + __expf(-z));         // SiLU gate
      yg[(size_t)row * D_INNER + d] = (_Float16)(y * zg);
    }
  }
}

// ---------------------------------------------------------------------------
extern "C" void kernel_launch(void* const* d_in, const int* in_sizes, int n_in,
                              void* d_out, int out_size, void* d_ws,
                              size_t ws_size, hipStream_t stream) {
  (void)in_sizes; (void)n_in; (void)out_size; (void)ws_size;
  const float* x      = (const float*)d_in[0];
  const float* ln_w   = (const float*)d_in[1];
  const float* ln_b   = (const float*)d_in[2];
  const float* W_in   = (const float*)d_in[3];
  const float* conv_w = (const float*)d_in[4];
  const float* conv_b = (const float*)d_in[5];
  const float* W_x    = (const float*)d_in[6];
  const float* W_dt   = (const float*)d_in[7];
  const float* b_dt   = (const float*)d_in[8];
  const float* A_log  = (const float*)d_in[9];
  const float* D_par  = (const float*)d_in[10];
  const float* W_out  = (const float*)d_in[11];
  float* out = (float*)d_out;

  char* ws = (char*)d_ws;
  size_t off = 0;
  auto carve = [&](size_t bytes) -> char* {
    char* p = ws + off;
    off += (bytes + 255) & ~(size_t)255;
    return p;
  };
  _Float16* xn16  = (_Float16*)carve((size_t)NROW * DIMC * 2);
  float*    xz    = (float*)carve((size_t)NROW * 2 * D_INNER * 4);
  float*    uf    = (float*)carve((size_t)NROW * D_INNER * 4);
  _Float16* uh    = (_Float16*)carve((size_t)NROW * D_INNER * 2);
  float*    xdbl  = (float*)carve((size_t)NROW * XCOLS * 4);
  float*    dl    = (float*)carve((size_t)NROW * D_INNER * 4);
  float*    Pb    = (float*)carve((size_t)NCHUNK * NBATCH * D_INNER * D_STATE * 4);
  float*    Qb    = (float*)carve((size_t)NCHUNK * NBATCH * D_INNER * D_STATE * 4);
  float*    Hin   = (float*)carve((size_t)NCHUNK * NBATCH * D_INNER * D_STATE * 4);
  _Float16* yg    = (_Float16*)carve((size_t)NROW * D_INNER * 2);
  _Float16* win16 = (_Float16*)carve((size_t)2 * D_INNER * DIMC * 2);
  _Float16* wx16  = (_Float16*)carve((size_t)XCOLS * D_INNER * 2);
  _Float16* wout16= (_Float16*)carve((size_t)DIMC * D_INNER * 2);

  // 0) weight prep (f32 -> f16, Bt layout = [N][K], W_x zero-padded to 48 rows)
  cvt_f16_kernel<<<(2 * D_INNER * DIMC + 255) / 256, 256, 0, stream>>>(
      W_in, win16, 2 * D_INNER * DIMC);
  pack_wx_kernel<<<(XCOLS * D_INNER + 255) / 256, 256, 0, stream>>>(W_x, wx16);
  cvt_f16_kernel<<<(DIMC * D_INNER + 255) / 256, 256, 0, stream>>>(
      W_out, wout16, DIMC * D_INNER);

  // 1) layernorm, (B,C,L) -> f16 (B*L, 128)
  ln_kernel<<<dim3(LSEQ / 32, NBATCH), 256, 0, stream>>>(x, ln_w, ln_b, xn16);

  // 2) xz = xn @ W_in^T : M=27648 N=512 K=128, 32x64 macro-tile per wave
  //    row strips = 27648/32 = 864, wavesPerRow = 512/64 = 8 -> 6912 waves
  gemm_wmma<2, 4, DIMC, false><<<(NROW / 32) * 8 / 8, 256, 0, stream>>>(
      xn16, win16, xz, NROW, 2 * D_INNER, 8);

  // 3) depthwise causal conv + SiLU -> u (f32 for scan, f16 for GEMMs)
  conv_silu_kernel<<<(NROW * D_INNER) / 256, 256, 0, stream>>>(
      xz, conv_w, conv_b, uf, uh);

  // 4) x_dbl = u @ W_x^T : M=27648 N=48(pad) K=256, 32x48 per wave -> 864 waves
  gemm_wmma<2, 3, D_INNER, false><<<(NROW / 32) / 8, 256, 0, stream>>>(
      uh, wx16, xdbl, NROW, XCOLS, 1);

  // 5) delta = softplus(dt_r @ W_dt^T + b_dt)
  delta_kernel<<<(NROW * D_INNER) / 256, 256, 0, stream>>>(xdbl, W_dt, b_dt, dl);

  // 6) chunked diagonal linear scan (3 passes)
  scan_pass1<<<dim3(NCHUNK, NBATCH, D_INNER / 16), 256, 0, stream>>>(
      dl, xdbl, uf, A_log, Pb, Qb);
  scan_pass2<<<dim3(NBATCH, D_INNER / 16), 256, 0, stream>>>(Pb, Qb, Hin);
  scan_pass3<<<dim3(NCHUNK, NBATCH, D_INNER / 16), 256, 0, stream>>>(
      dl, xdbl, uf, xz, A_log, D_par, Hin, yg);

  // 7) out = ygated @ W_out^T : N=128 K=256, transposed store to (B, C, L)
  //    wavesPerRow = 128/64 = 2 -> 1728 waves
  gemm_wmma<2, 4, D_INNER, true><<<(NROW / 32) * 2 / 8, 256, 0, stream>>>(
      yg, wout16, out, NROW, DIMC, 2);
}
